// V31Model_43250320670988
// MI455X (gfx1250) — matrix operand order repaired and checked by reading, not compiled, not run on hardware
//
#include <hip/hip_runtime.h>

// ---------------- model constants (from reference) ----------------
#define L_SEQ     128
#define DMODEL    1024
#define DINNER    2048
#define DSTATE    64
#define HEADDIM   64
#define NHEADS    32
#define DCONV     4
#define CONVDIM   2176            // DINNER + 2*DSTATE
#define DINPROJ   4256            // 2*DINNER + 2*DSTATE + NHEADS
#define NLAYERS   12
#define BASESPLIT 6
#define RANK      8
#define LORA_SC   2.0f
#define VOCAB     50277
#define NSTEPS    8
#define EPSF      1e-5f

typedef __attribute__((ext_vector_type(16))) __bf16 v16bf;
typedef __attribute__((ext_vector_type(8)))  float  v8f;

__device__ __forceinline__ float siluf(float x) { return x / (1.0f + expf(-x)); }

// gfx1250 async global->LDS copy (128b per lane), tracked by ASYNCcnt
__device__ __forceinline__ void async_b128(unsigned lds_off, const float* gaddr) {
  asm volatile("global_load_async_to_lds_b128 %0, %1, off"
               :: "v"(lds_off), "v"(gaddr) : "memory");
}

// =====================================================================
// WMMA GEMM:  Y[M,N] = X[M,K] @ W[N,K]^T   (f32 in/out, bf16 WMMA inside)
// Block: 256 threads = 8 waves. Tile: 16 (M) x 128 (N); each wave owns 16 N.
// Double-buffered LDS fed by global_load_async_to_lds_b128; ASYNCcnt pipelining.
// Requires: M % 16 == 0 (grid.y = M/16), K % 32 == 0.
// =====================================================================
#define AS_STRIDE 36                     // floats per LDS row (144B, 16B aligned)
#define AS_BUFB   (16  * AS_STRIDE * 4)  // bytes per A buffer
#define BS_BUFB   (128 * AS_STRIDE * 4)  // bytes per B buffer

__device__ __forceinline__ void issue_tile(const float* __restrict__ X,
                                           const float* __restrict__ W,
                                           int N, int K, int m0, int n0, int k0,
                                           unsigned asOff, unsigned bsOff, int tid) {
  // A tile 16x32 f32 = 128 b128 chunks; even lanes of every wave issue one
  // (=> every wave issues exactly 1 async instruction here)
  if ((tid & 1) == 0) {
    int id = tid >> 1;                 // 0..127
    int r = id >> 3, cg = id & 7;      // row, col-group(4 floats)
    const float* ga = X + (size_t)(m0 + r) * K + k0 + cg * 4;
    unsigned la = asOff + (unsigned)((r * AS_STRIDE + cg * 4) * 4);
    async_b128(la, ga);
  }
  // B tile 128x32 f32 = 1024 chunks; 4 async instructions per wave
#pragma unroll
  for (int it = 0; it < 4; ++it) {
    int id = tid + 256 * it;
    int r = id >> 3, cg = id & 7;
    int n = n0 + r; if (n > N - 1) n = N - 1;   // clamp tail (cols unstored)
    const float* gb = W + (size_t)n * K + k0 + cg * 4;
    unsigned lb = bsOff + (unsigned)((r * AS_STRIDE + cg * 4) * 4);
    async_b128(lb, gb);
  }
}

__global__ __launch_bounds__(256)
void wmma_gemm_xwt(const float* __restrict__ X, const float* __restrict__ W,
                   float* __restrict__ Y, int M, int N, int K) {
  __shared__ __align__(16) float As[2][16][AS_STRIDE];
  __shared__ __align__(16) float Bs[2][128][AS_STRIDE];
  const int tid    = threadIdx.x;
  const int wave   = tid >> 5;
  const int lane   = tid & 31;
  const int laneHi = lane >> 4;
  const int lm     = lane & 15;
  const int m0     = blockIdx.y * 16;
  const int n0     = blockIdx.x * 128;
  const unsigned asBase = (unsigned)(unsigned long long)(&As[0][0][0]);
  const unsigned bsBase = (unsigned)(unsigned long long)(&Bs[0][0][0]);
  const int ntiles = K >> 5;

  v8f c = {};

  // prologue: tile 0 -> buffer 0
  issue_tile(X, W, N, K, m0, n0, 0, asBase, bsBase, tid);

  for (int kt = 0; kt < ntiles; ++kt) {
    const int cur = kt & 1;
    if (kt + 1 < ntiles) {
      // overlap: issue next tile into the other buffer, then wait only for
      // the previous 5 async ops (per-wave, in-order completion)
      issue_tile(X, W, N, K, m0, n0, (kt + 1) << 5,
                 asBase + (unsigned)((cur ^ 1) * AS_BUFB),
                 bsBase + (unsigned)((cur ^ 1) * BS_BUFB), tid);
      asm volatile("s_wait_asynccnt 0x5" ::: "memory");
    } else {
      asm volatile("s_wait_asynccnt 0x0" ::: "memory");
    }
    __syncthreads();   // tile kt resident in buffer `cur` for all waves

    // Fragments per CDNA5 ISA 7.12.2 (16-bit A 16x32 / B 32x16 layouts)
    const float* arow = &As[cur][lm][0];
    const float* brow = &Bs[cur][wave * 16 + lm][0];
    v16bf a, b;
#pragma unroll
    for (int e = 0; e < 16; ++e) {
      int ka = e + ((e >= 8) ? 8 : 0) + (laneHi ? 8 : 0);
      a[e] = (__bf16)arow[ka];
      b[e] = (__bf16)brow[ka];
    }
    c = __builtin_amdgcn_wmma_f32_16x16x32_bf16(false, a, false, b,
                                                (short)0, c, false, false);
    __syncthreads();   // readers done before this buffer is overwritten
  }

  // C/D layout: VGPR r -> M=r (lanes 0-15) / M=r+8 (lanes 16-31), N=lane%16
  const int col = n0 + wave * 16 + lm;
  if (col < N) {
#pragma unroll
    for (int r = 0; r < 8; ++r) {
      int row = m0 + r + (laneHi ? 8 : 0);
      Y[(size_t)row * N + col] = c[r];
    }
  }
}

// =====================================================================
// lm_head GEMV: one wave per vocab row, float4 (b128) loads
// =====================================================================
__global__ __launch_bounds__(256)
void gemv_rows(const float* __restrict__ h, const float* __restrict__ W,
               float* __restrict__ out, int N, int K) {
  int row  = blockIdx.x * 8 + (threadIdx.x >> 5);
  int lane = threadIdx.x & 31;
  if (row >= N) return;
  const float4* wr = (const float4*)(W + (size_t)row * K);
  const float4* h4 = (const float4*)h;
  float s = 0.0f;
  for (int j = lane; j < (K >> 2); j += 32) {
    float4 a = h4[j], b = wr[j];
    s += a.x * b.x + a.y * b.y + a.z * b.z + a.w * b.w;
  }
#pragma unroll
  for (int o = 16; o > 0; o >>= 1) s += __shfl_xor(s, o);
  if (lane == 0) out[row] = s;
}

// =====================================================================
// RMSNorm over rows (in-place safe)
// =====================================================================
__global__ __launch_bounds__(256)
void rmsnorm_rows(const float* __restrict__ in, const float* __restrict__ w,
                  float* __restrict__ out, int D) {
  const float* row = in + (size_t)blockIdx.x * D;
  float* orow = out + (size_t)blockIdx.x * D;
  __shared__ float red[256];
  float s = 0.0f;
  for (int i = threadIdx.x; i < D; i += 256) { float v = row[i]; s += v * v; }
  red[threadIdx.x] = s; __syncthreads();
  for (int o = 128; o > 0; o >>= 1) {
    if (threadIdx.x < o) red[threadIdx.x] += red[threadIdx.x + o];
    __syncthreads();
  }
  float scale = rsqrtf(red[0] / (float)D + EPSF);
  for (int i = threadIdx.x; i < D; i += 256) orow[i] = row[i] * scale * w[i];
}

// y_out = rmsnorm(ys * silu(z), norm_w); z lives inside zx rows (stride DINPROJ)
__global__ __launch_bounds__(256)
void gate_rmsnorm(const float* __restrict__ ys, const float* __restrict__ zx,
                  const float* __restrict__ w, float* __restrict__ out) {
  const float* y = ys + (size_t)blockIdx.x * DINNER;
  const float* z = zx + (size_t)blockIdx.x * DINPROJ;
  float* o = out + (size_t)blockIdx.x * DINNER;
  __shared__ float red[256];
  float s = 0.0f;
  for (int i = threadIdx.x; i < DINNER; i += 256) {
    float v = y[i] * siluf(z[i]);
    o[i] = v;
    s += v * v;
  }
  red[threadIdx.x] = s; __syncthreads();
  for (int o2 = 128; o2 > 0; o2 >>= 1) {
    if (threadIdx.x < o2) red[threadIdx.x] += red[threadIdx.x + o2];
    __syncthreads();
  }
  float scale = rsqrtf(red[0] / (float)DINNER + EPSF);
  for (int i = threadIdx.x; i < DINNER; i += 256) o[i] = o[i] * scale * w[i];
}

// final h = rmsnorm(x[last] + res[last], norm_f_w)
__global__ __launch_bounds__(256)
void rms_last_token(const float* __restrict__ x, const float* __restrict__ res,
                    const float* __restrict__ w, float* __restrict__ out) {
  const float* xr = x + (size_t)(L_SEQ - 1) * DMODEL;
  const float* rr = res + (size_t)(L_SEQ - 1) * DMODEL;
  __shared__ float red[256];
  float s = 0.0f;
  for (int i = threadIdx.x; i < DMODEL; i += 256) { float v = xr[i] + rr[i]; s += v * v; }
  red[threadIdx.x] = s; __syncthreads();
  for (int o = 128; o > 0; o >>= 1) {
    if (threadIdx.x < o) red[threadIdx.x] += red[threadIdx.x + o];
    __syncthreads();
  }
  float scale = rsqrtf(red[0] / (float)DMODEL + EPSF);
  for (int i = threadIdx.x; i < DMODEL; i += 256) out[i] = (xr[i] + rr[i]) * scale * w[i];
}

// =====================================================================
// causal depthwise conv (width 4) + SiLU over zx[:, DINNER : DINNER+CONVDIM]
// =====================================================================
__global__ void conv_silu_k(const float* __restrict__ zx, const float* __restrict__ cw,
                            const float* __restrict__ cb, float* __restrict__ xbc) {
  int idx = blockIdx.x * 256 + threadIdx.x;
  if (idx >= L_SEQ * CONVDIM) return;
  int t = idx / CONVDIM, ch = idx % CONVDIM;
  float acc = cb[ch];
#pragma unroll
  for (int k = 0; k < DCONV; ++k) {
    int tt = t - (DCONV - 1) + k;
    if (tt >= 0) acc += zx[(size_t)tt * DINPROJ + DINNER + ch] * cw[ch * DCONV + k];
  }
  xbc[idx] = siluf(acc);
}

// dt = softplus(raw + dt_bias); dA = exp(-exp(A_log) * dt)
__global__ void dt_da_k(const float* __restrict__ zx, const float* __restrict__ dtb,
                        const float* __restrict__ Alog, float* __restrict__ dt,
                        float* __restrict__ dA) {
  int idx = blockIdx.x * 256 + threadIdx.x;
  if (idx >= L_SEQ * NHEADS) return;
  int t = idx >> 5, h = idx & 31;
  float raw = zx[(size_t)t * DINPROJ + DINNER + CONVDIM + h] + dtb[h];
  float sp = (raw > 20.0f) ? raw : log1pf(expf(raw));
  dt[idx] = sp;
  dA[idx] = expf(-expf(Alog[h]) * sp);
}

// =====================================================================
// selective scan: one block per head; h[64x64] in registers (16/thread)
// =====================================================================
__global__ __launch_bounds__(256)
void scan_k(const float* __restrict__ xbc, const float* __restrict__ dt,
            const float* __restrict__ dA, const float* __restrict__ Dv,
            float* __restrict__ y) {
  const int head = blockIdx.x;
  const int tid  = threadIdx.x;
  const int p    = tid >> 2;
  const int nq   = tid & 3;
  float h[16];
#pragma unroll
  for (int j = 0; j < 16; ++j) h[j] = 0.0f;
  const float Dh = Dv[head];
  for (int t = 0; t < L_SEQ; ++t) {
    const float* row = xbc + (size_t)t * CONVDIM;
    float xv  = row[head * HEADDIM + p];
    float dtv = dt[t * NHEADS + head];
    float dAv = dA[t * NHEADS + head];
    float dtx = dtv * xv;
    float acc = 0.0f;
#pragma unroll
    for (int j = 0; j < 16; ++j) {
      int n = nq * 16 + j;
      float Bn = row[DINNER + n];
      float Cn = row[DINNER + DSTATE + n];
      h[j] = h[j] * dAv + dtx * Bn;
      acc += h[j] * Cn;
    }
    acc += __shfl_xor(acc, 1);
    acc += __shfl_xor(acc, 2);
    if (nq == 0) y[(size_t)t * DINNER + head * HEADDIM + p] = acc + Dh * xv;
  }
}

// =====================================================================
// LoRA: T[t,r] = dot(X[t], A[r]);  Y[t,j] += 2 * sum_r T[t,r]*B[j,r]
// =====================================================================
__global__ void lora_down_k(const float* __restrict__ X, const float* __restrict__ A,
                            float* __restrict__ T, int K) {
  int idx = blockIdx.x * 256 + threadIdx.x;
  if (idx >= L_SEQ * RANK) return;
  int t = idx >> 3, r = idx & 7;
  const float* x = X + (size_t)t * K;
  const float* a = A + (size_t)r * K;
  float s = 0.0f;
  for (int j = 0; j < K; ++j) s += x[j] * a[j];
  T[idx] = s;
}

__global__ void lora_up_add_k(const float* __restrict__ T, const float* __restrict__ Bm,
                              float* __restrict__ Y, int N) {
  int idx = blockIdx.x * 256 + threadIdx.x;
  if (idx >= L_SEQ * N) return;
  int t = idx / N, j = idx % N;
  const float* tr = T + t * RANK;
  const float* br = Bm + (size_t)j * RANK;
  float s = 0.0f;
#pragma unroll
  for (int r = 0; r < RANK; ++r) s += tr[r] * br[r];
  Y[(size_t)t * N + j] += LORA_SC * s;
}

// ---------------- elementwise helpers ----------------
__global__ void embed_k(const int* __restrict__ ids, const float* __restrict__ emb,
                        float* __restrict__ x) {
  int idx = blockIdx.x * 256 + threadIdx.x;
  if (idx >= L_SEQ * DMODEL) return;
  int t = idx >> 10, d = idx & (DMODEL - 1);
  x[idx] = emb[(size_t)ids[t] * DMODEL + d];
}
__global__ void vec_copy_k(float* __restrict__ d, const float* __restrict__ a, int n) {
  int i = blockIdx.x * 256 + threadIdx.x;
  if (i < n) d[i] = a[i];
}
__global__ void vec_add_k(float* __restrict__ d, const float* __restrict__ a,
                          const float* __restrict__ b, int n) {
  int i = blockIdx.x * 256 + threadIdx.x;
  if (i < n) d[i] = a[i] + b[i];
}
__global__ void add_bcast_k(float* __restrict__ x, const float* __restrict__ v, int n) {
  int i = blockIdx.x * 256 + threadIdx.x;
  if (i < n) x[i] += v[i & (DMODEL - 1)];
}

// =====================================================================
// softmax + top-5 per step (one block per step)
// =====================================================================
__global__ __launch_bounds__(256)
void softmax_top5(const float* __restrict__ logits, float* __restrict__ top_v,
                  float* __restrict__ top_i) {
  const float* lg = logits + (size_t)blockIdx.x * VOCAB;
  __shared__ float rv[256];
  __shared__ int ri[256];
  __shared__ float s_max, s_sum;
  __shared__ int found[5];
  int tid = threadIdx.x;
  float m = -3.4e38f;
  for (int v = tid; v < VOCAB; v += 256) m = fmaxf(m, lg[v]);
  rv[tid] = m; __syncthreads();
  for (int o = 128; o > 0; o >>= 1) {
    if (tid < o) rv[tid] = fmaxf(rv[tid], rv[tid + o]);
    __syncthreads();
  }
  if (tid == 0) s_max = rv[0];
  __syncthreads();
  float gmax = s_max;
  float s = 0.0f;
  for (int v = tid; v < VOCAB; v += 256) s += expf(lg[v] - gmax);
  rv[tid] = s; __syncthreads();
  for (int o = 128; o > 0; o >>= 1) {
    if (tid < o) rv[tid] += rv[tid + o];
    __syncthreads();
  }
  if (tid == 0) s_sum = rv[0];
  __syncthreads();
  float gsum = s_sum;
  for (int pass = 0; pass < 5; ++pass) {
    float bv = -3.4e38f; int bi = 0x7fffffff;
    for (int v = tid; v < VOCAB; v += 256) {
      bool skip = false;
      for (int q = 0; q < pass; ++q) if (found[q] == v) skip = true;
      float val = lg[v];
      if (!skip && (val > bv || (val == bv && v < bi))) { bv = val; bi = v; }
    }
    rv[tid] = bv; ri[tid] = bi; __syncthreads();
    for (int o = 128; o > 0; o >>= 1) {
      if (tid < o) {
        if (rv[tid + o] > rv[tid] || (rv[tid + o] == rv[tid] && ri[tid + o] < ri[tid])) {
          rv[tid] = rv[tid + o]; ri[tid] = ri[tid + o];
        }
      }
      __syncthreads();
    }
    if (tid == 0) {
      found[pass] = ri[0];
      top_v[blockIdx.x * 5 + pass] = expf(rv[0] - gmax) / gsum;
      top_i[blockIdx.x * 5 + pass] = (float)ri[0];
    }
    __syncthreads();
  }
}

// =====================================================================
// Host orchestration
// =====================================================================
struct MixW {
  const float *Win, *Wout, *cw, *cb, *dtb, *Alog, *Dv, *nw;
  const float *inA, *inB, *outA, *outB;
};

struct Ws {
  float *x, *res, *xn, *zx, *xbc, *dt, *dA, *t1, *ys, *yn, *mo, *hl;
};

static inline dim3 g1(int n) { return dim3((n + 255) / 256); }

static void run_mixer(float* dst, const float* src, const MixW& m, const Ws& w,
                      hipStream_t s) {
  dim3 gin((DINPROJ + 127) / 128, L_SEQ / 16);
  wmma_gemm_xwt<<<gin, 256, 0, s>>>(src, m.Win, w.zx, L_SEQ, DINPROJ, DMODEL);
  if (m.inA) {
    lora_down_k<<<g1(L_SEQ * RANK), 256, 0, s>>>(src, m.inA, w.t1, DMODEL);
    lora_up_add_k<<<g1(L_SEQ * DINPROJ), 256, 0, s>>>(w.t1, m.inB, w.zx, DINPROJ);
  }
  conv_silu_k<<<g1(L_SEQ * CONVDIM), 256, 0, s>>>(w.zx, m.cw, m.cb, w.xbc);
  dt_da_k<<<g1(L_SEQ * NHEADS), 256, 0, s>>>(w.zx, m.dtb, m.Alog, w.dt, w.dA);
  scan_k<<<NHEADS, 256, 0, s>>>(w.xbc, w.dt, w.dA, m.Dv, w.ys);
  gate_rmsnorm<<<L_SEQ, 256, 0, s>>>(w.ys, w.zx, m.nw, w.yn);
  dim3 gout((DMODEL + 127) / 128, L_SEQ / 16);
  wmma_gemm_xwt<<<gout, 256, 0, s>>>(w.yn, m.Wout, dst, L_SEQ, DMODEL, DINNER);
  if (m.outA) {
    lora_down_k<<<g1(L_SEQ * RANK), 256, 0, s>>>(w.yn, m.outA, w.t1, DINNER);
    lora_up_add_k<<<g1(L_SEQ * DMODEL), 256, 0, s>>>(w.t1, m.outB, dst, DMODEL);
  }
}

static void run_block(bool first, const float* lnw, const MixW& m, const Ws& w,
                      hipStream_t s) {
  const int n = L_SEQ * DMODEL;
  if (first) vec_copy_k<<<g1(n), 256, 0, s>>>(w.res, w.x, n);
  else       vec_add_k<<<g1(n), 256, 0, s>>>(w.res, w.x, w.res, n);
  rmsnorm_rows<<<L_SEQ, 256, 0, s>>>(w.res, lnw, w.xn, DMODEL);
  run_mixer(w.x, w.xn, m, w, s);
}

extern "C" void kernel_launch(void* const* d_in, const int* in_sizes, int n_in,
                              void* d_out, int out_size, void* d_ws, size_t ws_size,
                              hipStream_t stream) {
  (void)in_sizes; (void)n_in; (void)out_size; (void)ws_size;
  int c = 0;
  const int* ids = (const int*)d_in[c++];
  c++;  // n (== 8 in the harness)
  auto getf = [&]() { return (const float*)d_in[c++]; };
  const float* embed    = getf();
  const float* lm_head  = getf();
  const float* step_emb = getf();
  const float* loop_nw  = getf();
  const float* norm_fw  = getf();
  auto fill = [&](MixW& m, bool lora) {
    m.Win = getf(); m.Wout = getf(); m.cw = getf(); m.cb = getf();
    m.dtb = getf(); m.Alog = getf(); m.Dv = getf(); m.nw = getf();
    if (lora) { m.inA = getf(); m.inB = getf(); m.outA = getf(); m.outB = getf(); }
    else { m.inA = m.inB = m.outA = m.outB = nullptr; }
  };
  MixW core; fill(core, false);
  const float* lnorm[NLAYERS]; MixW lay[NLAYERS];
  for (int i = 0; i < NLAYERS; ++i) { lnorm[i] = getf(); fill(lay[i], i >= BASESPLIT); }

  // workspace carve-out
  float* p = (float*)d_ws;
  Ws w;
  w.x   = p; p += L_SEQ * DMODEL;
  w.res = p; p += L_SEQ * DMODEL;
  w.xn  = p; p += L_SEQ * DMODEL;
  w.zx  = p; p += L_SEQ * DINPROJ;
  w.xbc = p; p += L_SEQ * CONVDIM;
  w.dt  = p; p += L_SEQ * NHEADS;
  w.dA  = p; p += L_SEQ * NHEADS;
  w.t1  = p; p += L_SEQ * RANK;
  w.ys  = p; p += L_SEQ * DINNER;
  w.yn  = p; p += L_SEQ * DINNER;
  w.mo  = p; p += L_SEQ * DMODEL;
  w.hl  = p; p += DMODEL;

  float* out = (float*)d_out;
  float* top_v = out + (size_t)NSTEPS * VOCAB;
  float* top_i = top_v + NSTEPS * 5;

  // ---- forward ----
  embed_k<<<g1(L_SEQ * DMODEL), 256, 0, stream>>>(ids, embed, w.x);
  for (int i = 0; i < NLAYERS; ++i)
    run_block(i == 0, lnorm[i], lay[i], w, stream);

  for (int step = 0; step < NSTEPS; ++step) {
    add_bcast_k<<<g1(L_SEQ * DMODEL), 256, 0, stream>>>(
        w.x, step_emb + (size_t)step * DMODEL, L_SEQ * DMODEL);
    for (int i = BASESPLIT; i < NLAYERS; ++i)
      run_block(false, lnorm[i], lay[i], w, stream);
    run_mixer(w.mo, w.x, core, w, stream);
    vec_add_k<<<g1(L_SEQ * DMODEL), 256, 0, stream>>>(w.x, w.x, w.mo, L_SEQ * DMODEL);
    rmsnorm_rows<<<L_SEQ, 256, 0, stream>>>(w.x, loop_nw, w.x, DMODEL);
    rms_last_token<<<1, 256, 0, stream>>>(w.x, w.res, norm_fw, w.hl);
    gemv_rows<<<(VOCAB + 7) / 8, 256, 0, stream>>>(
        w.hl, lm_head, out + (size_t)step * VOCAB, VOCAB, DMODEL);
  }
  softmax_top5<<<NSTEPS, 256, 0, stream>>>(out, top_v, top_i);
}
